// YOLOv2Head_68324339745215
// MI455X (gfx1250) — matrix operand order repaired and checked by reading, not compiled, not run on hardware
//
#include <hip/hip_runtime.h>
#include <hip/hip_bf16.h>

// ---------------------------------------------------------------------------
// YOLOv2 head on MI455X (gfx1250), wave32 + v_wmma_f32_16x16x32_bf16.
//
// conv1 (3x3, 512->1024, 32 imgs x 32x32): 3.09e11 FLOP -> WMMA implicit GEMM
// conv2 (1x1, 1024->425):                  2.85e10 FLOP -> WMMA GEMM
// decode (~164K anchors of sigmoid/exp/softmax-80)      -> VALU
//
// HBM traffic ~260MB -> ~11us @ 23.3TB/s; the matrix-core time dominates, so
// both convs run bf16 operands / f32 accumulate. Features are packed into a
// ZERO-PADDED [B][34][34][512] bf16 tensor so every conv1 A-fragment load is
// unconditional (no exec-mask branches in the hot loop), and each wave holds
// a 2Mx2N register tile (4 independent accumulators) so WMMA->WMMA hazard
// slots are filled with independent WMMAs and loads amortize 2 per WMMA.
// ---------------------------------------------------------------------------

#define SSZ   32
#define SP    34               // padded spatial (halo of 1 on each side)
#define CIN   512
#define C1    1024
#define K1    (9*CIN)          // 4608
#define BATCH 32
#define NCLS  80
#define NANC  5
#define COUT  425              // 5*(5+80)
#define COUTP 432              // padded to 27*16
#define MTOT  (BATCH*SSZ*SSZ)  // 32768

typedef __attribute__((ext_vector_type(16))) __bf16 v16bf;
typedef __attribute__((ext_vector_type(8)))  float  v8f;

union ABf { uint4 u[2]; v16bf v; };   // 32 bytes = 16 bf16 = one WMMA operand

__constant__ float c_AW[NANC] = {42.0f, 98.0f, 180.0f, 300.0f, 400.0f};
__constant__ float c_AH[NANC] = {45.0f, 130.0f, 260.0f, 180.0f, 400.0f};

static __device__ __forceinline__ unsigned short f32_to_bf16(float f) {
    unsigned int u = __float_as_uint(f);
    u += 0x7fffu + ((u >> 16) & 1u);      // round-to-nearest-even
    return (unsigned short)(u >> 16);
}

// ---------------------------------------------------------------------------
// Pack kernels (one-shot layout conversions, HBM-bandwidth trivial)
// ---------------------------------------------------------------------------

// features (B,512,32,32) f32 NCHW -> Ftp[b][y+1][x+1][c] bf16 with zero halo.
__global__ __launch_bounds__(256) void pack_feat_pad(const float* __restrict__ f,
                                                     unsigned short* __restrict__ Ftp) {
    int o = blockIdx.x * 256 + threadIdx.x;          // < 18,939,904
    int c  = o & (CIN - 1);
    int p  = o >> 9;
    int xp = p % SP;
    int yp = (p / SP) % SP;
    int b  = p / (SP * SP);
    unsigned short v = 0;
    if (xp >= 1 && xp <= SSZ && yp >= 1 && yp <= SSZ) {
        v = f32_to_bf16(f[(((size_t)(b * CIN + c)) * SSZ + (yp - 1)) * SSZ + (xp - 1)]);
    }
    Ftp[o] = v;
}

// conv1_w (1024,512,3,3) OIHW f32 -> Wb[cout][ (dy*3+dx)*512 + c ] bf16
__global__ __launch_bounds__(256) void pack_w1(const float* __restrict__ w,
                                               unsigned short* __restrict__ Wb) {
    int o = blockIdx.x * 256 + threadIdx.x;          // < 4,718,592
    int k    = o % K1;
    int cout = o / K1;
    int tap = k >> 9;
    int c   = k & (CIN - 1);
    int dy = tap / 3, dx = tap % 3;
    Wb[o] = f32_to_bf16(w[(((size_t)cout * CIN + c) * 3 + dy) * 3 + dx]);
}

// conv2_w (425,1024,1,1) -> W2b[co<432][n] bf16, zero-padded rows 425..431
__global__ __launch_bounds__(256) void pack_w2(const float* __restrict__ w,
                                               unsigned short* __restrict__ W2b) {
    int o = blockIdx.x * 256 + threadIdx.x;          // < 442,368
    int n  = o & (C1 - 1);
    int co = o >> 10;
    float v = (co < COUT) ? w[(size_t)co * C1 + n] : 0.0f;
    W2b[o] = f32_to_bf16(v);
}

// BN fold: scale = gamma/sqrt(var+eps); shift = beta - mean*scale; pad bias2
__global__ __launch_bounds__(256) void bn_prep(const float* __restrict__ g,
                                               const float* __restrict__ be,
                                               const float* __restrict__ mu,
                                               const float* __restrict__ va,
                                               const float* __restrict__ b2,
                                               float* __restrict__ scale,
                                               float* __restrict__ shift,
                                               float* __restrict__ bias2) {
    int i = blockIdx.x * 256 + threadIdx.x;          // 4 blocks -> i < 1024
    if (i < C1) {
        float sc = g[i] * rsqrtf(va[i] + 1e-5f);
        scale[i] = sc;
        shift[i] = be[i] - mu[i] * sc;
    }
    if (i < COUTP) bias2[i] = (i < COUT) ? b2[i] : 0.0f;
}

// ---------------------------------------------------------------------------
// conv1 3x3 implicit-GEMM + BN + leaky ReLU, bf16 WMMA, f32 accumulate.
// Block: 256 thr = 8 waves covering M=32 (full x row) x N=256 channels.
// Wave register tile: 2M x 2N = four 16x16 f32 accumulators.
// Grid: x = 4 channel slabs, y = 32 rows, z = 32 images.
// ---------------------------------------------------------------------------
__global__ __launch_bounds__(256) void conv1_wmma(
    const unsigned short* __restrict__ Ftp,  // [B][34][34][512] bf16, padded
    const unsigned short* __restrict__ Wb,   // [1024][4608] bf16
    const float* __restrict__ scale,
    const float* __restrict__ shift,
    unsigned short* __restrict__ Xa) {       // [B][32][32][1024] bf16
    const int tid  = threadIdx.x;
    const int wave = tid >> 5;
    const int lane = tid & 31;
    const int half = lane >> 4;
    const int lid  = lane & 15;

    const int y = blockIdx.y;
    const int b = blockIdx.z;
    const int nbase = blockIdx.x * 256 + wave * 32;

    const int co0 = nbase + lid;
    const int co1 = nbase + 16 + lid;
    const unsigned short* __restrict__ Wrow0 = Wb + (size_t)co0 * K1;
    const unsigned short* __restrict__ Wrow1 = Wb + (size_t)co1 * K1;

    v8f acc00 = {}; v8f acc01 = {};          // (M-tile 0) x (N-tile 0/1)
    v8f acc10 = {}; v8f acc11 = {};          // (M-tile 1) x (N-tile 0/1)

    for (int tap = 0; tap < 9; ++tap) {
        const int dy = tap / 3, dx = tap % 3;
        // Padded coords: yp = y+dy in [0,33], xp = x+lid+dx in [0,33].
        const size_t rb = ((size_t)(b * SP + (y + dy))) * SP;
        const unsigned short* __restrict__ Arow0 = Ftp + (rb + (lid + dx)) * CIN;
        const unsigned short* __restrict__ Arow1 = Arow0 + (size_t)16 * CIN;

        for (int c0 = 0; c0 < CIN; c0 += 32) {
            const int k0 = tap * CIN + c0;
            // A frags: 16x32 bf16 (ISA 7.12.2), lane half -> K {h*8..h*8+7}
            // and {16+h*8..}; unconditional 16B loads (zero halo in Ftp).
            ABf a0, a1;
            a0.u[0] = *reinterpret_cast<const uint4*>(Arow0 + c0 + half * 8);
            a0.u[1] = *reinterpret_cast<const uint4*>(Arow0 + c0 + 16 + half * 8);
            a1.u[0] = *reinterpret_cast<const uint4*>(Arow1 + c0 + half * 8);
            a1.u[1] = *reinterpret_cast<const uint4*>(Arow1 + c0 + 16 + half * 8);
            // B frags: 32x16 bf16, column = cout, lane half -> K {h*8..h*8+7} and {16+h*8..}.
            ABf b0, b1;
            b0.u[0] = *reinterpret_cast<const uint4*>(Wrow0 + k0 + half * 8);
            b0.u[1] = *reinterpret_cast<const uint4*>(Wrow0 + k0 + 16 + half * 8);
            b1.u[0] = *reinterpret_cast<const uint4*>(Wrow1 + k0 + half * 8);
            b1.u[1] = *reinterpret_cast<const uint4*>(Wrow1 + k0 + 16 + half * 8);

            acc00 = __builtin_amdgcn_wmma_f32_16x16x32_bf16(
                false, a0.v, false, b0.v, (short)0, acc00, false, false);
            acc01 = __builtin_amdgcn_wmma_f32_16x16x32_bf16(
                false, a0.v, false, b1.v, (short)0, acc01, false, false);
            acc10 = __builtin_amdgcn_wmma_f32_16x16x32_bf16(
                false, a1.v, false, b0.v, (short)0, acc10, false, false);
            acc11 = __builtin_amdgcn_wmma_f32_16x16x32_bf16(
                false, a1.v, false, b1.v, (short)0, acc11, false, false);
        }
    }

    // Epilogue: BN + leaky ReLU, store bf16 activation (conv2's A matrix).
    const float sc0 = scale[co0], sh0 = shift[co0];
    const float sc1 = scale[co1], sh1 = shift[co1];
    const size_t rowbase = ((size_t)(b * SSZ + y)) * SSZ;
#pragma unroll
    for (int r = 0; r < 8; ++r) {
        const int m = half * 8 + r;          // C layout: M = half*8+r, N = lid
        unsigned short* __restrict__ o0 = Xa + (rowbase + m) * C1;        // x = m
        unsigned short* __restrict__ o1 = Xa + (rowbase + 16 + m) * C1;   // x = 16+m
        float v;
        v = acc00[r] * sc0 + sh0; v = v > 0.0f ? v : 0.1f * v; o0[co0] = f32_to_bf16(v);
        v = acc01[r] * sc1 + sh1; v = v > 0.0f ? v : 0.1f * v; o0[co1] = f32_to_bf16(v);
        v = acc10[r] * sc0 + sh0; v = v > 0.0f ? v : 0.1f * v; o1[co0] = f32_to_bf16(v);
        v = acc11[r] * sc1 + sh1; v = v > 0.0f ? v : 0.1f * v; o1[co1] = f32_to_bf16(v);
    }
}

// ---------------------------------------------------------------------------
// conv2 1x1 GEMM: M=32768, K=1024, N=432(pad). Wave = 2 M-tiles x 1 N-tile
// (B-fragment reused across both WMMAs, 2 independent accumulators).
// Grid: x = 1024 (32 rows each), y = 4 (8 N-tiles per block, 27 used).
// ---------------------------------------------------------------------------
__global__ __launch_bounds__(256) void conv2_wmma(
    const unsigned short* __restrict__ Xa,   // [32768][1024] bf16
    const unsigned short* __restrict__ W2b,  // [432][1024] bf16
    const float* __restrict__ bias2,         // [432]
    float* __restrict__ Preds) {             // [32768][432] f32
    const int wave = threadIdx.x >> 5;
    const int lane = threadIdx.x & 31;
    const int half = lane >> 4;
    const int lid  = lane & 15;
    const int mt0 = blockIdx.x * 2;          // two consecutive M-tiles
    const int nt  = blockIdx.y * 8 + wave;
    if (nt >= COUTP / 16) return;            // wave-uniform exit

    const int co = nt * 16 + lid;
    const unsigned short* __restrict__ Arow0 = Xa + ((size_t)(mt0 * 16 + lid)) * C1;
    const unsigned short* __restrict__ Arow1 = Arow0 + (size_t)16 * C1;
    const unsigned short* __restrict__ Brow  = W2b + (size_t)co * C1;

    v8f acc0 = {}; v8f acc1 = {};
    for (int k0 = 0; k0 < C1; k0 += 32) {
        ABf a0, a1, bb;
        a0.u[0] = *reinterpret_cast<const uint4*>(Arow0 + k0 + half * 8);
        a0.u[1] = *reinterpret_cast<const uint4*>(Arow0 + k0 + 16 + half * 8);
        a1.u[0] = *reinterpret_cast<const uint4*>(Arow1 + k0 + half * 8);
        a1.u[1] = *reinterpret_cast<const uint4*>(Arow1 + k0 + 16 + half * 8);
        bb.u[0] = *reinterpret_cast<const uint4*>(Brow + k0 + half * 8);
        bb.u[1] = *reinterpret_cast<const uint4*>(Brow + k0 + 16 + half * 8);
        acc0 = __builtin_amdgcn_wmma_f32_16x16x32_bf16(
            false, a0.v, false, bb.v, (short)0, acc0, false, false);
        acc1 = __builtin_amdgcn_wmma_f32_16x16x32_bf16(
            false, a1.v, false, bb.v, (short)0, acc1, false, false);
    }

    const float bv = bias2[co];
#pragma unroll
    for (int r = 0; r < 8; ++r) {
        const int m = half * 8 + r;
        Preds[(size_t)(mt0 * 16 + m) * COUTP + co]        = acc0[r] + bv;
        Preds[(size_t)((mt0 + 1) * 16 + m) * COUTP + co]  = acc1[r] + bv;
    }
}

// ---------------------------------------------------------------------------
// Decode: one thread per (b, y, x, anchor). Stable softmax-80, argmax, boxes.
// out5 [B][5120][5] f32 followed by labels [B][5120] (stored as float).
// ---------------------------------------------------------------------------
__global__ __launch_bounds__(256) void decode_kernel(
    const float* __restrict__ Preds, float* __restrict__ out) {
    const int tid = blockIdx.x * 256 + threadIdx.x;  // < 163840
    const int a   = tid % NANC;
    const int pos = (tid / NANC) & 1023;             // (y*32+x)
    const int b   = tid / (NANC * 1024);
    const int y = pos >> 5, x = pos & 31;

    const float* __restrict__ p = Preds + ((size_t)(b * 1024 + pos)) * COUTP + a * 85;
    const float tx = p[0], ty = p[1], tw = p[2], th = p[3], to = p[4];
    const float sx  = 1.0f / (1.0f + expf(-tx));
    const float sy  = 1.0f / (1.0f + expf(-ty));
    const float bw  = expf(fminf(tw, 8.0f));
    const float bh  = expf(fminf(th, 8.0f));
    const float obj = 1.0f / (1.0f + expf(-to));

    float best = p[5];
    int arg = 0;
    for (int j = 1; j < NCLS; ++j) {
        const float v = p[5 + j];
        if (v > best) { best = v; arg = j; }         // first-occurrence argmax
    }
    float sum = 0.0f;
    for (int j = 0; j < NCLS; ++j) sum += expf(p[5 + j] - best);
    const float score = obj * (1.0f / sum);          // obj * max(softmax)

    const float cx = (sx + (float)x) * 32.0f;
    const float cy = (sy + (float)y) * 32.0f;
    const float pw = c_AW[a] * bw;                   // (anc/stride)*bw*stride
    const float ph = c_AH[a] * bh;
    const float x1 = fminf(fmaxf(cx - 0.5f * pw, 0.0f), 1023.0f);
    const float y1 = fminf(fmaxf(cy - 0.5f * ph, 0.0f), 1023.0f);
    const float x2 = fminf(fmaxf(cx + 0.5f * pw, 0.0f), 1023.0f);
    const float y2 = fminf(fmaxf(cy + 0.5f * ph, 0.0f), 1023.0f);

    const int i = pos * NANC + a;
    float* __restrict__ o = out + ((size_t)b * 5120 + i) * 5;
    o[0] = x1; o[1] = y1; o[2] = x2; o[3] = y2; o[4] = score;
    out[(size_t)BATCH * 5120 * 5 + (size_t)b * 5120 + i] = (float)arg;
}

// ---------------------------------------------------------------------------
extern "C" void kernel_launch(void* const* d_in, const int* in_sizes, int n_in,
                              void* d_out, int out_size, void* d_ws, size_t ws_size,
                              hipStream_t stream) {
    const float* features = (const float*)d_in[0];
    const float* conv1_w  = (const float*)d_in[1];
    const float* bn_gamma = (const float*)d_in[2];
    const float* bn_beta  = (const float*)d_in[3];
    const float* bn_mean  = (const float*)d_in[4];
    const float* bn_var   = (const float*)d_in[5];
    const float* conv2_w  = (const float*)d_in[6];
    const float* conv2_b  = (const float*)d_in[7];
    float* out = (float*)d_out;

    // Workspace layout (bytes), all 16B aligned; total ~172 MB.
    char* ws = (char*)d_ws;
    const size_t off_Ftp   = 0;                                        // 37,879,808
    const size_t off_Wb    = off_Ftp   + (size_t)BATCH * SP * SP * CIN * 2;
    const size_t off_W2b   = off_Wb    + (size_t)C1 * K1 * 2;          // +9,437,184
    const size_t off_scale = off_W2b   + (size_t)COUTP * C1 * 2;       // +884,736
    const size_t off_shift = off_scale + 4096;
    const size_t off_bias2 = off_shift + 4096;
    const size_t off_Xa    = off_bias2 + 2048;
    const size_t off_Preds = off_Xa    + (size_t)MTOT * C1 * 2;        // +67,108,864

    unsigned short* Ftp   = (unsigned short*)(ws + off_Ftp);
    unsigned short* Wb    = (unsigned short*)(ws + off_Wb);
    unsigned short* W2b   = (unsigned short*)(ws + off_W2b);
    float*          scale = (float*)(ws + off_scale);
    float*          shift = (float*)(ws + off_shift);
    float*          bias2 = (float*)(ws + off_bias2);
    unsigned short* Xa    = (unsigned short*)(ws + off_Xa);
    float*          Preds = (float*)(ws + off_Preds);

    bn_prep<<<4, 256, 0, stream>>>(bn_gamma, bn_beta, bn_mean, bn_var, conv2_b,
                                   scale, shift, bias2);
    pack_feat_pad<<<73984, 256, 0, stream>>>(features, Ftp);   // 18,939,904 elts
    pack_w1<<<18432, 256, 0, stream>>>(conv1_w, Wb);
    pack_w2<<<1728, 256, 0, stream>>>(conv2_w, W2b);

    conv1_wmma<<<dim3(4, 32, 32), 256, 0, stream>>>(Ftp, Wb, scale, shift, Xa);
    conv2_wmma<<<dim3(1024, 4, 1), 256, 0, stream>>>(Xa, W2b, bias2, Preds);
    decode_kernel<<<640, 256, 0, stream>>>(Preds, out);
}
